// HybridModel_82712480187017
// MI455X (gfx1250) — matrix operand order repaired and checked by reading, not compiled, not run on hardware
//
#include <hip/hip_runtime.h>
#include <math.h>

typedef __attribute__((ext_vector_type(16))) _Float16 v16h;
typedef __attribute__((ext_vector_type(8)))  float    v8f;
typedef __attribute__((ext_vector_type(2)))  float    v2f;

// Fused hybrid model:
//   angles = x@Wp^T + bp            -> V_WMMA_F32_16X16X32_F16 (f16 in, f32 acc)
//   psi    = U' * r(angles)         -> V_WMMA_F32_16X16X4_F32  (full f32)
//   z      = signed prob sums       -> Hadamard-factored VALU
//   out    = W2 relu(W1 z + b1)     -> per-lane VALU
// U' is the batch-independent circuit unitary with per-column phases folded in,
// built once per block by 16 threads (trivial cost).
__global__ __launch_bounds__(256, 1) void fused_hybrid(
    const float* __restrict__ x,   // [B,64]
    const float* __restrict__ Wp,  // [4,64]
    const float* __restrict__ bp,  // [4]
    const float* __restrict__ wq,  // [2,4,3]
    const float* __restrict__ W1,  // [64,4]
    const float* __restrict__ b1,  // [64]
    const float* __restrict__ W2,  // [1,64]
    const float* __restrict__ b2,  // [1]
    float* __restrict__ out,       // [B,1]
    int B)
{
  __shared__ float2 sU[256];          // U'[k][j]
  __shared__ float  sW1[256];         // [64][4]
  __shared__ float  sb1[64];
  __shared__ float  sW2[64];
  __shared__ float  sBp[4];
  __shared__ float  sB2;
  __shared__ float  sAng[8 * 32 * 4]; // per-wave 32 rows x 4 angles
  __shared__ float  sS[8 * 32 * 16];  // per-wave 32 rows x 16 probabilities

  const int tid  = threadIdx.x;
  const int wave = tid >> 5;          // wave32
  const int lane = tid & 31;
  const int n    = lane & 15;         // column index in WMMA B/C/D layouts
  const int h    = lane >> 4;         // lane half

  // ---- stage classifier weights ----
  sW1[tid] = W1[tid];
  if (tid < 64) { sb1[tid] = b1[tid]; sW2[tid] = W2[tid]; }
  if (tid < 4)  sBp[tid] = bp[tid];
  if (tid == 0) sB2 = b2[0];

  // ---- build U' : column j = circuit(|j>), then * (-i)^popcount(j) ----
  if (tid < 16) {
    const int j = tid;
    float ar[16], ai[16];
    #pragma unroll
    for (int k = 0; k < 16; ++k) { ar[k] = 0.f; ai[k] = 0.f; }
    ar[j] = 1.f;
    #pragma unroll
    for (int layer = 0; layer < 2; ++layer) {
      #pragma unroll
      for (int w = 0; w < 4; ++w) {
        const int bit = 3 - w;
        const float t0 = wq[(layer*4 + w)*3 + 0];
        const float t1 = wq[(layer*4 + w)*3 + 1];
        const float t2 = wq[(layer*4 + w)*3 + 2];
        { // RZ(t0)
          float sh, ch; sincosf(0.5f*t0, &sh, &ch);
          #pragma unroll
          for (int k = 0; k < 16; ++k) {
            float s = ((k>>bit)&1) ? sh : -sh;
            float r = ar[k], im = ai[k];
            ar[k] = r*ch - im*s;
            ai[k] = r*s  + im*ch;
          }
        }
        { // RY(t1)
          float sh, ch; sincosf(0.5f*t1, &sh, &ch);
          #pragma unroll
          for (int k = 0; k < 16; ++k)
            if (((k>>bit)&1) == 0) {
              int k1 = k + (1<<bit);
              float a0r=ar[k], a0i=ai[k], a1r=ar[k1], a1i=ai[k1];
              ar[k]  = ch*a0r - sh*a1r;  ai[k]  = ch*a0i - sh*a1i;
              ar[k1] = sh*a0r + ch*a1r;  ai[k1] = sh*a0i + ch*a1i;
            }
        }
        { // RZ(t2)
          float sh, ch; sincosf(0.5f*t2, &sh, &ch);
          #pragma unroll
          for (int k = 0; k < 16; ++k) {
            float s = ((k>>bit)&1) ? sh : -sh;
            float r = ar[k], im = ai[k];
            ar[k] = r*ch - im*s;
            ai[k] = r*s  + im*ch;
          }
        }
      }
      #pragma unroll
      for (int c = 0; c < 3; ++c) {   // CNOT(c, c+1)
        const int bc = 3 - c, bt = 3 - (c + 1);
        #pragma unroll
        for (int k = 0; k < 16; ++k)
          if (((k>>bc)&1) == 1 && ((k>>bt)&1) == 0) {
            int k1 = k | (1<<bt);
            float tr = ar[k]; ar[k] = ar[k1]; ar[k1] = tr;
            float ti = ai[k]; ai[k] = ai[k1]; ai[k1] = ti;
          }
      }
    }
    const int p = __popc(j) & 3;      // (-i)^p
    const float pr = (p==0) ? 1.f : ((p==2) ? -1.f : 0.f);
    const float pi = (p==1) ? -1.f : ((p==3) ? 1.f : 0.f);
    #pragma unroll
    for (int k = 0; k < 16; ++k) {
      float2 u;
      u.x = ar[k]*pr - ai[k]*pi;
      u.y = ar[k]*pi + ai[k]*pr;
      sU[k*16 + j] = u;
    }
  }

  // ---- B-matrix for angles GEMM: Wp^T padded to N=16, f16 WMMA layout ----
  v16h bmat0, bmat1;
  #pragma unroll
  for (int e = 0; e < 16; ++e) {
    const int K = (e & 7) + ((e >> 3) << 4) + 8*h;
    const float w0 = (n < 4) ? Wp[n*64 + K]      : 0.f;
    const float w1 = (n < 4) ? Wp[n*64 + K + 32] : 0.f;
    bmat0[e] = (_Float16)w0;
    bmat1[e] = (_Float16)w1;
  }

  __syncthreads();

  // ---- B fragments for psi GEMM: Re/Im of U'^T in f32 WMMA layout ----
  // B is 4x16 (KxN) per chunk: lane holds N=n, K = 4*kc + 2h + e
  v2f bre[4], bim[4];
  #pragma unroll
  for (int kc = 0; kc < 4; ++kc) {
    #pragma unroll
    for (int e = 0; e < 2; ++e) {
      float2 u = sU[n*16 + 4*kc + 2*h + e];
      bre[kc][e] = u.x;
      bim[kc][e] = u.y;
    }
  }
  const float bpn = sBp[n & 3];       // bias for this lane's output column

  const long rowsBase = (long)blockIdx.x * 256 + (long)wave * 32;

  // ---- angles via 2x V_WMMA_F32_16X16X32_F16 per 16-row tile ----
  #pragma unroll
  for (int T = 0; T < 2; ++T) {
    long aRow = rowsBase + T*16 + n;            // A row M = lane%16
    if (aRow >= B) aRow = B - 1;                // clamp keeps loads in-bounds
    const float* xr = x + aRow * 64;
    v16h amat0, amat1;
    #pragma unroll
    for (int p = 0; p < 8; ++p) {               // element pairs (e=2p, 2p+1)
      const int Kb = 2*p + ((p & 4) << 1) + 8*h;
      float2 v0 = *(const float2*)(xr + Kb);
      float2 v1 = *(const float2*)(xr + Kb + 32);
      amat0[2*p]   = (_Float16)v0.x;
      amat0[2*p+1] = (_Float16)v0.y;
      amat1[2*p]   = (_Float16)v1.x;
      amat1[2*p+1] = (_Float16)v1.y;
    }
    v8f acc = {};
    acc = __builtin_amdgcn_wmma_f32_16x16x32_f16(false, amat0, false, bmat0,
                                                 (short)0, acc, false, false);
    acc = __builtin_amdgcn_wmma_f32_16x16x32_f16(false, amat1, false, bmat1,
                                                 (short)0, acc, false, false);
    if (n < 4) {                                // D: lane=col N, vgpr r -> row 8h+r
      #pragma unroll
      for (int r = 0; r < 8; ++r)
        sAng[(wave*32 + T*16 + 8*h + r)*4 + n] = acc[r] + bpn;
    }
  }

  __syncthreads();

  // ---- psi = U' * r via V_WMMA_F32_16X16X4_F32 (re and im planes) ----
  // A fragment is generated directly in layout: lane owns tile row n,
  // element (kc, e) of the product-state vector = r01[kc] * r23[2h+e].
  #pragma unroll
  for (int T = 0; T < 2; ++T) {
    const float* ap = &sAng[(wave*32 + T*16 + n)*4];
    float cc[4], ss[4];
    #pragma unroll
    for (int i = 0; i < 4; ++i) sincosf(0.5f*ap[i], &ss[i], &cc[i]);
    float r01[4];
    r01[0]=cc[0]*cc[1]; r01[1]=cc[0]*ss[1]; r01[2]=ss[0]*cc[1]; r01[3]=ss[0]*ss[1];
    const float f2  = h ? ss[2] : cc[2];
    const float rb0 = f2*cc[3];                 // r23[2h+0]
    const float rb1 = f2*ss[3];                 // r23[2h+1]

    v8f pre = {}, pim = {};
    #pragma unroll
    for (int kc = 0; kc < 4; ++kc) {
      v2f a;
      a[0] = r01[kc]*rb0;
      a[1] = r01[kc]*rb1;
      pre = __builtin_amdgcn_wmma_f32_16x16x4_f32(false, a, false, bre[kc],
                                                  (short)0, pre, false, false);
      pim = __builtin_amdgcn_wmma_f32_16x16x4_f32(false, a, false, bim[kc],
                                                  (short)0, pim, false, false);
    }
    // probabilities; D layout: lane holds amplitude index k=n, rows 8h+r
    #pragma unroll
    for (int r = 0; r < 8; ++r) {
      const float prr = pre[r], pii = pim[r];
      sS[(wave*32 + T*16 + 8*h + r)*16 + n] = prr*prr + pii*pii;
    }
  }

  __syncthreads();

  // ---- per-lane: one batch row; z via factored signed sums, then MLP ----
  const long row = rowsBase + lane;
  if (row < B) {
    const float* pp = &sS[(wave*32 + lane)*16];
    float p[16];
    #pragma unroll
    for (int k = 0; k < 16; ++k) p[k] = pp[k];

    // z_i = sum_k (-1)^{bit_(3-i)(k)} p_k, Hadamard-factored
    float s1[8], d1[8];
    #pragma unroll
    for (int jj = 0; jj < 8; ++jj) { s1[jj]=p[2*jj]+p[2*jj+1]; d1[jj]=p[2*jj]-p[2*jj+1]; }
    float s2[4], d2[4];
    #pragma unroll
    for (int jj = 0; jj < 4; ++jj) { s2[jj]=s1[2*jj]+s1[2*jj+1]; d2[jj]=s1[2*jj]-s1[2*jj+1]; }
    float z[4];
    z[3] = ((d1[0]+d1[1])+(d1[2]+d1[3])) + ((d1[4]+d1[5])+(d1[6]+d1[7]));
    z[2] = (d2[0]+d2[1]) + (d2[2]+d2[3]);
    const float s30 = s2[0]+s2[1], s31 = s2[2]+s2[3];
    z[1] = (s2[0]-s2[1]) + (s2[2]-s2[3]);
    z[0] = s30 - s31;

    float o = sB2;
    #pragma unroll 8
    for (int jj = 0; jj < 64; ++jj) {
      float hj = sb1[jj];
      hj = fmaf(sW1[jj*4+0], z[0], hj);
      hj = fmaf(sW1[jj*4+1], z[1], hj);
      hj = fmaf(sW1[jj*4+2], z[2], hj);
      hj = fmaf(sW1[jj*4+3], z[3], hj);
      hj = fmaxf(hj, 0.f);
      o  = fmaf(sW2[jj], hj, o);
    }
    out[row] = o;
  }
}

extern "C" void kernel_launch(void* const* d_in, const int* in_sizes, int n_in,
                              void* d_out, int out_size, void* d_ws, size_t ws_size,
                              hipStream_t stream) {
  const float* x  = (const float*)d_in[0];
  const float* Wp = (const float*)d_in[1];
  const float* bp = (const float*)d_in[2];
  const float* wq = (const float*)d_in[3];
  const float* W1 = (const float*)d_in[4];
  const float* b1 = (const float*)d_in[5];
  const float* W2 = (const float*)d_in[6];
  const float* b2 = (const float*)d_in[7];
  float* out = (float*)d_out;

  const int B = in_sizes[0] / 64;              // x is [B,64]
  const int grid = (B + 255) / 256;            // 256 rows per block (8 waves x 32)
  fused_hybrid<<<grid, 256, 0, stream>>>(x, Wp, bp, wq, W1, b1, W2, b2, out, B);
}
